// GNN_node_61993557950823
// MI455X (gfx1250) — compile-verified, matrix operand order checked by reading, not run on hardware
//
#include <hip/hip_runtime.h>

#define Nn 200000
#define Ee 400000
#define Dd 128
#define Ll 5
#define AVc 100
#define BVc 10
#define BN_EPS 1e-5f

typedef __attribute__((ext_vector_type(16))) _Float16 v16h;
typedef __attribute__((ext_vector_type(8)))  _Float16 v8h;
typedef __attribute__((ext_vector_type(8)))  float    v8f;

// ---------------------------------------------------------------------------
// Encoders
// ---------------------------------------------------------------------------
__global__ void encode_nodes(const int* __restrict__ x, const float* __restrict__ at,
                             float* __restrict__ h, float* __restrict__ total) {
    const int n = blockIdx.x;
    const int d = threadIdx.x;
    float s = 0.f;
#pragma unroll
    for (int c = 0; c < 9; ++c) {
        const int idx = x[n * 9 + c];
        s += at[((size_t)c * AVc + idx) * Dd + d];
    }
    h[(size_t)n * Dd + d] = s;
    total[(size_t)n * Dd + d] = s;   // total starts as h
}

__global__ void encode_edges(const int* __restrict__ ea, const float* __restrict__ bt,
                             float* __restrict__ eemb) {
    const int e = blockIdx.x;
    const int d = threadIdx.x;
    float s = 0.f;
#pragma unroll
    for (int c = 0; c < 3; ++c) {
        const int idx = ea[e * 3 + c];
        s += bt[((size_t)c * BVc + idx) * Dd + d];
    }
    eemb[(size_t)e * Dd + d] = s;
}

// ---------------------------------------------------------------------------
// Degree / normalization
// ---------------------------------------------------------------------------
__global__ void deg_init(float* __restrict__ deg) {
    const int n = blockIdx.x * blockDim.x + threadIdx.x;
    if (n < Nn) deg[n] = 1.0f;
}

__global__ void deg_accum(const int* __restrict__ row, float* __restrict__ deg) {
    const int e = blockIdx.x * blockDim.x + threadIdx.x;
    if (e < Ee) atomicAdd(&deg[row[e]], 1.0f);
}

__global__ void deg_finish(const float* __restrict__ deg, float* __restrict__ dis,
                           float* __restrict__ dinv) {
    const int n = blockIdx.x * blockDim.x + threadIdx.x;
    if (n < Nn) {
        const float dv = deg[n];
        dis[n]  = rsqrtf(dv);
        dinv[n] = 1.0f / dv;
    }
}

__global__ void edge_norm(const int* __restrict__ row, const int* __restrict__ col,
                          const float* __restrict__ dis, float* __restrict__ nv) {
    const int e = blockIdx.x * blockDim.x + threadIdx.x;
    if (e < Ee) nv[e] = dis[row[e]] * dis[col[e]];
}

// ---------------------------------------------------------------------------
// One-time weight conversion: all 5 layers of W (f32 [out,in]) -> f16
// ---------------------------------------------------------------------------
__global__ void convert_w(const float* __restrict__ W, _Float16* __restrict__ w16) {
    const int i = blockIdx.x * blockDim.x + threadIdx.x;   // Ll*Dd*Dd = 81920
    w16[i] = (_Float16)W[i];
}

// ---------------------------------------------------------------------------
// WMMA GEMM: hl[N,128] = h[N,128] @ W^T + bias   (W pre-converted to f16)
// 32 rows per block (2 M-tiles); 8 waves, one 16-col tile each; K = 4 x 32.
// B fragments loaded once per K-chunk and reused for both M-tiles.
// Fragment layout per ISA: lane half h in {0,1} holds the two contiguous
// 8-half runs {K = 32k + 8h .. +7} and {K = 32k + 16 + 8h .. +7}.
// ---------------------------------------------------------------------------
__global__ void __launch_bounds__(256) gemm_wmma_f16(const float* __restrict__ h,
                                                     const _Float16* __restrict__ w16,
                                                     const float* __restrict__ bl,
                                                     float* __restrict__ out) {
    __shared__ __align__(16) _Float16 As[32 * Dd];   // 8 KB A-tile (32 rows)
    const int tid = threadIdx.x;
    const int rowbase = blockIdx.x * 32;

    // Cooperative coalesced stage of the 32x128 A tile, f32 -> f16
#pragma unroll
    for (int j = 0; j < 16; ++j) {
        const int idx = tid + 256 * j;
        As[idx] = (_Float16)h[(size_t)rowbase * Dd + idx];
    }
    __syncthreads();

    const int wave = tid >> 5;                 // wave32: 8 waves per block
    const int lane = tid & 31;
    const int half = lane >> 4;                // lane group 0..15 vs 16..31
    const int mr   = lane & 15;                // row (A) / col (B,D) within tile
    const int nt   = wave;                     // output column tile
    const _Float16* Wrow = w16 + (size_t)(nt * 16 + mr) * Dd;

    v8f acc0 = {};
    v8f acc1 = {};
#pragma unroll
    for (int kc = 0; kc < 4; ++kc) {
        const int kb = kc * 32;
        // B fragment: two contiguous 16-byte runs from f16 weights
        const v8h b0 = *(const v8h*)(Wrow + kb + half * 8);
        const v8h b1 = *(const v8h*)(Wrow + kb + 16 + half * 8);
        v16h bfr;
#pragma unroll
        for (int i = 0; i < 8; ++i) { bfr[i] = b0[i]; bfr[i + 8] = b1[i]; }

        // A fragments for both M-tiles: same run structure, from LDS
#pragma unroll
        for (int mt = 0; mt < 2; ++mt) {
            const _Float16* Arow = As + (mt * 16 + mr) * Dd;
            const v8h a0 = *(const v8h*)(Arow + kb + half * 8);
            const v8h a1 = *(const v8h*)(Arow + kb + 16 + half * 8);
            v16h afr;
#pragma unroll
            for (int i = 0; i < 8; ++i) { afr[i] = a0[i]; afr[i + 8] = a1[i]; }
            if (mt == 0)
                acc0 = __builtin_amdgcn_wmma_f32_16x16x32_f16(
                    false, afr, false, bfr, (short)0, acc0, false, false);
            else
                acc1 = __builtin_amdgcn_wmma_f32_16x16x32_f16(
                    false, afr, false, bfr, (short)0, acc1, false, false);
        }
    }

    const float bias = bl[nt * 16 + mr];
#pragma unroll
    for (int r = 0; r < 8; ++r) {
        const int m = r + 8 * half;            // ISA C/D layout
        out[(size_t)(rowbase + m) * Dd + nt * 16 + mr]      = acc0[r] + bias;
        out[(size_t)(rowbase + 16 + m) * Dd + nt * 16 + mr] = acc1[r] + bias;
    }
}

// ---------------------------------------------------------------------------
// Per-layer elementwise / scatter / BN kernels
// ---------------------------------------------------------------------------
__global__ void self_term(const float* __restrict__ hl, const float* __restrict__ rootl,
                          const float* __restrict__ dinv, float* __restrict__ outb) {
    const size_t i = (size_t)blockIdx.x * blockDim.x + threadIdx.x;
    const int d = (int)(i & (Dd - 1));
    const int n = (int)(i >> 7);
    outb[i] = fmaxf(hl[i] + rootl[d], 0.0f) * dinv[n];
}

__global__ void scatter_msg(const int* __restrict__ row, const int* __restrict__ col,
                            const float* __restrict__ nv, const float* __restrict__ hl,
                            const float* __restrict__ eemb, float* __restrict__ outb) {
    const int e = blockIdx.x;
    const int d = threadIdx.x;
    const int r = row[e], c = col[e];
    const float m = nv[e] * fmaxf(hl[(size_t)r * Dd + d] + eemb[(size_t)e * Dd + d], 0.0f);
    atomicAdd(&outb[(size_t)c * Dd + d], m);
}

__global__ void zero_stats(float* __restrict__ stats) {
    stats[threadIdx.x] = 0.0f;   // 256 threads: sum[128] + sumsq[128]
}

// 2500 blocks x 80 rows; thread owns feature d = tid&127, strides rows by 2.
__global__ void __launch_bounds__(256) bn_stats(const float* __restrict__ outb,
                                                float* __restrict__ stats) {
    __shared__ float sm[256];
    __shared__ float sq[256];
    const int tid = threadIdx.x;
    const int d   = tid & (Dd - 1);
    const int sub = tid >> 7;                   // 0 or 1
    const int base = blockIdx.x * 80;
    float s = 0.f, q = 0.f;
    for (int r = sub; r < 80; r += 2) {
        const float v = outb[(size_t)(base + r) * Dd + d];
        s += v;
        q += v * v;
    }
    sm[tid] = s; sq[tid] = q;
    __syncthreads();
    if (tid < Dd) {
        s = sm[tid] + sm[tid + 128];
        q = sq[tid] + sq[tid + 128];
        atomicAdd(&stats[d], s);
        atomicAdd(&stats[Dd + d], q);
    }
}

__global__ void bn_finish(float* __restrict__ stats, const float* __restrict__ gammal,
                          const float* __restrict__ betal) {
    const int d = threadIdx.x;
    const float inv_n = 1.0f / (float)Nn;
    const float mean = stats[d] * inv_n;
    const float var  = stats[Dd + d] * inv_n - mean * mean;
    const float sc   = gammal[d] * rsqrtf(var + BN_EPS);
    stats[2 * Dd + d] = sc;
    stats[3 * Dd + d] = betal[d] - mean * sc;
}

__global__ void bn_apply(const float* __restrict__ outb, const float* __restrict__ stats,
                         float* __restrict__ total, float* __restrict__ hprev,
                         const int do_relu) {
    const size_t i = (size_t)blockIdx.x * blockDim.x + threadIdx.x;
    const int d = (int)(i & (Dd - 1));
    float v = outb[i] * stats[2 * Dd + d] + stats[3 * Dd + d];
    if (do_relu) v = fmaxf(v, 0.0f);
    total[i] += v;
    hprev[i] = v;
}

// ---------------------------------------------------------------------------
// Launch
// ---------------------------------------------------------------------------
extern "C" void kernel_launch(void* const* d_in, const int* in_sizes, int n_in,
                              void* d_out, int out_size, void* d_ws, size_t ws_size,
                              hipStream_t stream) {
    const int*   x           = (const int*)d_in[0];
    const int*   edge_index  = (const int*)d_in[1];
    const int*   edge_attr   = (const int*)d_in[2];
    const float* atom_tables = (const float*)d_in[3];
    const float* bond_tables = (const float*)d_in[4];
    const float* W           = (const float*)d_in[5];
    const float* b           = (const float*)d_in[6];
    const float* root        = (const float*)d_in[7];
    const float* gamma       = (const float*)d_in[8];
    const float* beta        = (const float*)d_in[9];
    float* total = (float*)d_out;

    // Workspace layout (floats)
    float* ws     = (float*)d_ws;
    float* h_prev = ws;                               // N*D
    float* hl     = h_prev + (size_t)Nn * Dd;         // N*D
    float* outb   = hl     + (size_t)Nn * Dd;         // N*D
    float* eemb   = outb   + (size_t)Nn * Dd;         // E*D
    float* deg    = eemb   + (size_t)Ee * Dd;         // N
    float* dis    = deg  + Nn;                        // N
    float* dinv   = dis  + Nn;                        // N
    float* nv     = dinv + Nn;                        // E
    float* stats  = nv   + Ee;                        // 4*D
    _Float16* w16 = (_Float16*)(stats + 4 * Dd);      // L*D*D halves

    const int* row = edge_index;
    const int* col = edge_index + Ee;

    // Encoders + init
    encode_nodes<<<Nn, Dd, 0, stream>>>(x, atom_tables, h_prev, total);
    encode_edges<<<Ee, Dd, 0, stream>>>(edge_attr, bond_tables, eemb);

    // Degree + symmetric norm
    deg_init  <<<(Nn + 255) / 256, 256, 0, stream>>>(deg);
    deg_accum <<<(Ee + 255) / 256, 256, 0, stream>>>(row, deg);
    deg_finish<<<(Nn + 255) / 256, 256, 0, stream>>>(deg, dis, dinv);
    edge_norm <<<(Ee + 255) / 256, 256, 0, stream>>>(row, col, dis, nv);

    // One-time f16 weight conversion (all layers)
    convert_w<<<(Ll * Dd * Dd) / 256, 256, 0, stream>>>(W, w16);

    const int elem_blocks = (Nn * Dd) / 256;          // 100000, exact

    for (int l = 0; l < Ll; ++l) {
        const _Float16* Wl16 = w16 + (size_t)l * Dd * Dd;
        const float*    bl   = b   + (size_t)l * Dd;

        gemm_wmma_f16<<<Nn / 32, 256, 0, stream>>>(h_prev, Wl16, bl, hl);
        self_term   <<<elem_blocks, 256, 0, stream>>>(hl, root + (size_t)l * Dd, dinv, outb);
        scatter_msg <<<Ee, Dd, 0, stream>>>(row, col, nv, hl, eemb, outb);

        zero_stats<<<1, 256, 0, stream>>>(stats);
        bn_stats  <<<Nn / 80, 256, 0, stream>>>(outb, stats);
        bn_finish <<<1, Dd, 0, stream>>>(stats, gamma + (size_t)l * Dd, beta + (size_t)l * Dd);
        bn_apply  <<<elem_blocks, 256, 0, stream>>>(outb, stats, total, h_prev,
                                                    (l < Ll - 1) ? 1 : 0);
    }
}
